// CQAttention_10084583211043
// MI455X (gfx1250) — matrix operand order repaired and checked by reading, not compiled, not run on hardware
//
#include <hip/hip_runtime.h>
#include <hip/hip_bf16.h>

// CQAttention for MI455X (gfx1250, wave32, WMMA f32_16x16x32_f16).
// B=64, H=256, LC=2048, LQ=256. Workspace requirement: ~96 MB.

#define B_   64
#define H_   256
#define LC_  2048
#define LQ_  256
#define NBLK 16            // LC/128
#define NEGV (-1e30f)

typedef __attribute__((ext_vector_type(16))) _Float16 v16h;
typedef __attribute__((ext_vector_type(8)))  float    v8f;
typedef __attribute__((ext_vector_type(4)))  float    v4f;

// ---- WMMA helpers -----------------------------------------------------------
__device__ __forceinline__ v8f wmma_f16(v16h a, v16h b, v8f c) {
  // D = A(16x32 f16) x B(32x16 f16) + C(16x16 f32)
  return __builtin_amdgcn_wmma_f32_16x16x32_f16(false, a, false, b, (short)0, c,
                                                false, false);
}

// A-matrix fragment (16x32 f16): lane m = lane&15, kbase = 16*(lane>>4);
// holds k = kbase..kbase+15 (contiguous 32B).
__device__ __forceinline__ v16h load_a_frag(const _Float16* row) {
  int lane = threadIdx.x & 31;
  int kb = (lane >> 4) << 4;          // 0 or 16
  v16h f;
  v4f* fp = reinterpret_cast<v4f*>(&f);
  fp[0] = *reinterpret_cast<const v4f*>(row + kb);
  fp[1] = *reinterpret_cast<const v4f*>(row + kb + 8);
  return f;
}

// B-matrix fragment (32x16 f16): lane n = lane&15, k = 16*(lane>>4)..+15
// (contiguous 32B).
__device__ __forceinline__ v16h load_b_frag(const _Float16* row) {
  int lane = threadIdx.x & 31;
  int ko = (lane >> 4) << 4;          // 0 or 16
  v16h f;
  v4f* fp = reinterpret_cast<v4f*>(&f);
  fp[0] = *reinterpret_cast<const v4f*>(row + ko);
  fp[1] = *reinterpret_cast<const v4f*>(row + ko + 8);
  return f;
}

// B fragment built from f32 memory (convert to f16 on the fly).
__device__ __forceinline__ v16h load_b_frag_f32(const float* row) {
  int lane = threadIdx.x & 31;
  int ko = (lane >> 4) << 4;
  v16h f;
#pragma unroll
  for (int t = 0; t < 16; ++t) f[t] = (_Float16)row[ko + t];
  return f;
}

// Online softmax merge: (m,s) <- merge((m,s),(m2,s2))
__device__ __forceinline__ void smerge(float& m, float& s, float m2, float s2) {
  float mm = fmaxf(m, m2);
  s = s * __expf(m - mm) + s2 * __expf(m2 - mm);
  m = mm;
}

// ---- K0a: transpose/convert C -> CW[b][i][h] = C[b][h][i]*w3[h] (f16), rowterm
__global__ void __launch_bounds__(256)
k_prep_c(const float* __restrict__ C, const float* __restrict__ lp,
         _Float16* __restrict__ CW, float* __restrict__ rowterm) {
  __shared__ float tile[64][65];
  int blk = blockIdx.x;
  int b  = blk >> 5;                  // LC/64 = 32 blocks per batch
  int i0 = (blk & 31) << 6;
  int t = threadIdx.x;
  const float* w1 = lp;
  const float* w3 = lp + 2 * H_;
  float rt = 0.f;
  for (int hc = 0; hc < 4; ++hc) {
    __syncthreads();
#pragma unroll
    for (int rep = 0; rep < 16; ++rep) {
      int hl = rep * 4 + (t >> 6);
      int il = t & 63;
      tile[il][hl] = C[((size_t)b * H_ + hc * 64 + hl) * LC_ + i0 + il];
    }
    __syncthreads();
#pragma unroll
    for (int rep = 0; rep < 16; ++rep) {
      int il = rep * 4 + (t >> 6);
      int hl = t & 63;
      int h = hc * 64 + hl;
      CW[((size_t)b * LC_ + i0 + il) * H_ + h] = (_Float16)(tile[il][hl] * w3[h]);
    }
    if (t < 64) {
      for (int hl = 0; hl < 64; ++hl) rt += tile[t][hl] * w1[hc * 64 + hl];
    }
  }
  if (t < 64) rowterm[(size_t)b * LC_ + i0 + t] = rt;
}

// ---- K0b: QT16[b][j][h], Q16[b][h][j] (f16), colterm -----------------------
__global__ void __launch_bounds__(256)
k_prep_q(const float* __restrict__ Q, const float* __restrict__ lp,
         _Float16* __restrict__ QT16, _Float16* __restrict__ Q16,
         float* __restrict__ colterm) {
  __shared__ float tile[64][65];
  int b = blockIdx.x;
  int t = threadIdx.x;
  const float* w2 = lp + H_;
  for (int idx = t; idx < H_ * LQ_; idx += 256)
    Q16[(size_t)b * H_ * LQ_ + idx] = (_Float16)Q[(size_t)b * H_ * LQ_ + idx];
  float ct[4] = {0.f, 0.f, 0.f, 0.f};
  for (int jc = 0; jc < 4; ++jc) {
    for (int hc = 0; hc < 4; ++hc) {
      __syncthreads();
#pragma unroll
      for (int rep = 0; rep < 16; ++rep) {
        int hl = rep * 4 + (t >> 6);
        int jl = t & 63;
        tile[jl][hl] = Q[((size_t)b * H_ + hc * 64 + hl) * LQ_ + jc * 64 + jl];
      }
      __syncthreads();
#pragma unroll
      for (int rep = 0; rep < 16; ++rep) {
        int jl = rep * 4 + (t >> 6);
        int hl = t & 63;
        QT16[((size_t)b * LQ_ + jc * 64 + jl) * H_ + hc * 64 + hl] =
            (_Float16)tile[jl][hl];
      }
      if (t < 64) {
        for (int hl = 0; hl < 64; ++hl) ct[jc] += tile[t][hl] * w2[hc * 64 + hl];
      }
    }
  }
  if (t < 64) {
    for (int jc = 0; jc < 4; ++jc)
      colterm[(size_t)b * LQ_ + jc * 64 + t] = ct[jc];
  }
}

// ---- K1: softmax statistics (row max/sum, column partial max/sum) ----------
__global__ void __launch_bounds__(256)
k_stats(const _Float16* __restrict__ CW, const _Float16* __restrict__ QT16,
        const float* __restrict__ rowterm, const float* __restrict__ colterm,
        const float* __restrict__ cmask, const float* __restrict__ qmask,
        float* __restrict__ rowmax, float* __restrict__ rowsum,
        float* __restrict__ cpmax, float* __restrict__ cpsum) {
  __shared__ float cmS[8][LQ_];
  __shared__ float csS[8][LQ_];
  int blk = blockIdx.x;
  int b  = blk >> 4;                  // NBLK = 16
  int ib = blk & 15;
  int i0 = ib * 128;
  int wave = threadIdx.x >> 5;
  int lane = threadIdx.x & 31;
  int half = lane >> 4;
  int n = lane & 15;
  int iS = i0 + wave * 16;

  v16h a[8];
  const _Float16* arow = CW + ((size_t)b * LC_ + iS + n) * H_;
#pragma unroll
  for (int kc = 0; kc < 8; ++kc) a[kc] = load_a_frag(arow + kc * 32);

  float rt[8], cm[8], runm[8], runs[8];
#pragma unroll
  for (int r = 0; r < 8; ++r) {
    int i = iS + half * 8 + r;
    rt[r] = rowterm[(size_t)b * LC_ + i];
    cm[r] = cmask[(size_t)b * LC_ + i];
    runm[r] = -3.0e38f;
    runs[r] = 0.f;
  }

  for (int jt = 0; jt < 16; ++jt) {
    int j = jt * 16 + n;
    v8f acc = {};
    const _Float16* brow = QT16 + ((size_t)b * LQ_ + j) * H_;
#pragma unroll
    for (int kc = 0; kc < 8; ++kc)
      acc = wmma_f16(a[kc], load_b_frag(brow + kc * 32), acc);
    float ctm = colterm[(size_t)b * LQ_ + j];
    float qm  = qmask[(size_t)b * LQ_ + j];
    float tmx = -3.0e38f, tss = 0.f;
    float lcv[8];
#pragma unroll
    for (int r = 0; r < 8; ++r) {
      float logit = acc[r] + rt[r] + ctm;
      float lr = logit * (1.f - qm) + qm * NEGV;   // row-masked logit
      float nm = fmaxf(runm[r], lr);
      runs[r] = runs[r] * __expf(runm[r] - nm) + __expf(lr - nm);
      runm[r] = nm;
      float lc = logit * (1.f - cm[r]) + cm[r] * NEGV;  // col-masked logit
      lcv[r] = lc;
      tmx = fmaxf(tmx, lc);
    }
#pragma unroll
    for (int r = 0; r < 8; ++r) tss += __expf(lcv[r] - tmx);
    // combine the two lane-halves (same column n)
    float om = __shfl_xor(tmx, 16);
    float os = __shfl_xor(tss, 16);
    smerge(tmx, tss, om, os);
    if (half == 0) { cmS[wave][j] = tmx; csS[wave][j] = tss; }
  }
  // reduce row stats across the 16 lanes of each half
#pragma unroll
  for (int r = 0; r < 8; ++r) {
#pragma unroll
    for (int m = 1; m < 16; m <<= 1) {
      float om = __shfl_xor(runm[r], m);
      float os = __shfl_xor(runs[r], m);
      smerge(runm[r], runs[r], om, os);
    }
  }
  if (n == 0) {
#pragma unroll
    for (int r = 0; r < 8; ++r) {
      int i = iS + half * 8 + r;
      rowmax[(size_t)b * LC_ + i] = runm[r];
      rowsum[(size_t)b * LC_ + i] = runs[r];
    }
  }
  __syncthreads();
  int j = threadIdx.x;
  float m = -3.0e38f, s = 0.f;
  for (int w = 0; w < 8; ++w) smerge(m, s, cmS[w][j], csS[w][j]);
  cpmax[((size_t)b * NBLK + ib) * LQ_ + j] = m;
  cpsum[((size_t)b * NBLK + ib) * LQ_ + j] = s;
}

// ---- K2: reduce column partials -------------------------------------------
__global__ void __launch_bounds__(256)
k_colreduce(const float* __restrict__ cpmax, const float* __restrict__ cpsum,
            float* __restrict__ colmax, float* __restrict__ colsum) {
  int b = blockIdx.x;
  int j = threadIdx.x;
  float m = -3.0e38f, s = 0.f;
  for (int blk = 0; blk < NBLK; ++blk)
    smerge(m, s, cpmax[((size_t)b * NBLK + blk) * LQ_ + j],
           cpsum[((size_t)b * NBLK + blk) * LQ_ + j]);
  colmax[(size_t)b * LQ_ + j] = m;
  colsum[(size_t)b * LQ_ + j] = s;
}

// ---- K3: T = S_col^T @ Ct, stored transposed f16: Tt[b][h][j] --------------
__global__ void __launch_bounds__(256)
k_T(const float* __restrict__ C, const _Float16* __restrict__ CW,
    const _Float16* __restrict__ QT16, const float* __restrict__ rowterm,
    const float* __restrict__ colterm, const float* __restrict__ cmask,
    const float* __restrict__ colmax, const float* __restrict__ colsum,
    _Float16* __restrict__ Tt) {
  __shared__ _Float16 Sc[64][40];     // exp(S_col) chunk, [j][i], padded
  int b  = blockIdx.x >> 2;
  int j0 = (blockIdx.x & 3) * 64;
  int wave = threadIdx.x >> 5;
  int lane = threadIdx.x & 31;
  int half = lane >> 4;
  int n = lane & 15;
  // S-phase tile assignment: 2 i-tiles x 4 j-tiles
  int itw = wave >> 2;
  int jtw = wave & 3;
  int jS = j0 + jtw * 16 + n;
  float ctm = colterm[(size_t)b * LQ_ + jS];
  float cmx = colmax[(size_t)b * LQ_ + jS];
  // T-phase assignment: 4 j-strips x 2 h-halves
  int jt2 = wave >> 1;
  int hh  = wave & 1;
  v8f zero = {};
  v8f accT[8];
#pragma unroll
  for (int t8 = 0; t8 < 8; ++t8) accT[t8] = zero;

  for (int ic = 0; ic < 64; ++ic) {
    int i0c = ic * 32;
    int iS = i0c + itw * 16;
    // recompute S tile -> exp(col-masked - colmax) -> Sc (f16, [j][i])
    v8f acc = {};
    const _Float16* arow = CW + ((size_t)b * LC_ + iS + n) * H_;
    const _Float16* brow = QT16 + ((size_t)b * LQ_ + jS) * H_;
#pragma unroll
    for (int kc = 0; kc < 8; ++kc)
      acc = wmma_f16(load_a_frag(arow + kc * 32), load_b_frag(brow + kc * 32),
                     acc);
#pragma unroll
    for (int r = 0; r < 8; ++r) {
      int i = iS + half * 8 + r;
      float logit = acc[r] + rowterm[(size_t)b * LC_ + i] + ctm;
      float cmr = cmask[(size_t)b * LC_ + i];
      float lc = logit * (1.f - cmr) + cmr * NEGV;
      float e = __expf(lc - cmx);
      Sc[jtw * 16 + n][itw * 16 + half * 8 + r] = (_Float16)e;
    }
    __syncthreads();
    // T accumulation: A = Sc rows (j-strip), B = Ct from raw C ([h][i] major)
    v16h aT = load_a_frag(&Sc[jt2 * 16 + n][0]);
#pragma unroll
    for (int t8 = 0; t8 < 8; ++t8) {
      int hg = hh * 128 + t8 * 16 + n;
      const float* crow = C + ((size_t)b * H_ + hg) * LC_ + i0c;
      accT[t8] = wmma_f16(aT, load_b_frag_f32(crow), accT[t8]);
    }
    __syncthreads();
  }
  float cs[8];
#pragma unroll
  for (int r = 0; r < 8; ++r)
    cs[r] = colsum[(size_t)b * LQ_ + j0 + jt2 * 16 + half * 8 + r];
#pragma unroll
  for (int t8 = 0; t8 < 8; ++t8) {
    int hg = hh * 128 + t8 * 16 + n;
#pragma unroll
    for (int r = 0; r < 8; ++r) {
      int j = j0 + jt2 * 16 + half * 8 + r;
      Tt[((size_t)b * H_ + hg) * LQ_ + j] = (_Float16)(accT[t8][r] / cs[r]);
    }
  }
}

// ---- K4: S_row recompute, A = S_row@Qt, Bt = S_row@T, fused output ---------
__global__ void __launch_bounds__(256)
k_out(const float* __restrict__ C, const _Float16* __restrict__ CW,
      const _Float16* __restrict__ QT16, const _Float16* __restrict__ Q16,
      const _Float16* __restrict__ Tt, const float* __restrict__ rowterm,
      const float* __restrict__ colterm, const float* __restrict__ qmask,
      const float* __restrict__ rowmax, const float* __restrict__ rowsum,
      float* __restrict__ out) {
  extern __shared__ _Float16 Srow[];  // [128][264] f16
  const int STR = 264;
  int b  = blockIdx.x >> 4;
  int i0 = (blockIdx.x & 15) * 128;
  int wave = threadIdx.x >> 5;
  int lane = threadIdx.x & 31;
  int half = lane >> 4;
  int n = lane & 15;
  int iS = i0 + wave * 16;

  // Phase 1: recompute S strip, normalize to S_row, stage as f16 in LDS
  v16h a[8];
  const _Float16* arow = CW + ((size_t)b * LC_ + iS + n) * H_;
#pragma unroll
  for (int kc = 0; kc < 8; ++kc) a[kc] = load_a_frag(arow + kc * 32);
  float rt[8], rm[8], rs[8];
#pragma unroll
  for (int r = 0; r < 8; ++r) {
    int i = iS + half * 8 + r;
    rt[r] = rowterm[(size_t)b * LC_ + i];
    rm[r] = rowmax[(size_t)b * LC_ + i];
    rs[r] = rowsum[(size_t)b * LC_ + i];
  }
  for (int jt = 0; jt < 16; ++jt) {
    int j = jt * 16 + n;
    v8f acc = {};
    const _Float16* brow = QT16 + ((size_t)b * LQ_ + j) * H_;
#pragma unroll
    for (int kc = 0; kc < 8; ++kc)
      acc = wmma_f16(a[kc], load_b_frag(brow + kc * 32), acc);
    float ctm = colterm[(size_t)b * LQ_ + j];
    float qm  = qmask[(size_t)b * LQ_ + j];
#pragma unroll
    for (int r = 0; r < 8; ++r) {
      float logit = acc[r] + rt[r] + ctm;
      float lr = logit * (1.f - qm) + qm * NEGV;
      float sv = __expf(lr - rm[r]) / rs[r];
      Srow[(size_t)(wave * 16 + half * 8 + r) * STR + j] = (_Float16)sv;
    }
  }
  __syncthreads();

  // Phase 2: A-fragments from LDS S_row; shared by both matmuls.
  v16h aS[8];
  const _Float16* srow = Srow + (size_t)(wave * 16 + n) * STR;
#pragma unroll
  for (int kc = 0; kc < 8; ++kc) aS[kc] = load_a_frag(srow + kc * 32);
  int ibase = iS + half * 8;
  for (int ht = 0; ht < 16; ++ht) {
    int h = ht * 16 + n;
    v8f accA = {}, accB = {};
    const _Float16* qrow = Q16 + ((size_t)b * H_ + h) * LQ_;
    const _Float16* trow = Tt  + ((size_t)b * H_ + h) * LQ_;
#pragma unroll
    for (int kc = 0; kc < 8; ++kc) {
      accA = wmma_f16(aS[kc], load_b_frag(qrow + kc * 32), accA);
      accB = wmma_f16(aS[kc], load_b_frag(trow + kc * 32), accB);
    }
    const float* crow = C + ((size_t)b * H_ + h) * LC_ + ibase;
    v4f c0 = *reinterpret_cast<const v4f*>(crow);
    v4f c1 = *reinterpret_cast<const v4f*>(crow + 4);
    float cv[8] = {c0[0], c0[1], c0[2], c0[3], c1[0], c1[1], c1[2], c1[3]};
    size_t ob = (size_t)b * (4 * H_) * LC_;
    float* p0 = out + ob + (size_t)(0 * H_ + h) * LC_ + ibase;
    float* p1 = out + ob + (size_t)(1 * H_ + h) * LC_ + ibase;
    float* p2 = out + ob + (size_t)(2 * H_ + h) * LC_ + ibase;
    float* p3 = out + ob + (size_t)(3 * H_ + h) * LC_ + ibase;
    v4f va0, va1, vb0, vb1, vc0, vc1;
#pragma unroll
    for (int r = 0; r < 4; ++r) {
      va0[r] = accA[r];            va1[r] = accA[4 + r];
      vb0[r] = cv[r] * accA[r];    vb1[r] = cv[4 + r] * accA[4 + r];
      vc0[r] = cv[r] * accB[r];    vc1[r] = cv[4 + r] * accB[4 + r];
    }
    *reinterpret_cast<v4f*>(p0) = c0;  *reinterpret_cast<v4f*>(p0 + 4) = c1;
    *reinterpret_cast<v4f*>(p1) = va0; *reinterpret_cast<v4f*>(p1 + 4) = va1;
    *reinterpret_cast<v4f*>(p2) = vb0; *reinterpret_cast<v4f*>(p2 + 4) = vb1;
    *reinterpret_cast<v4f*>(p3) = vc0; *reinterpret_cast<v4f*>(p3 + 4) = vc1;
  }
}

// ---- host launcher ----------------------------------------------------------
extern "C" void kernel_launch(void* const* d_in, const int* in_sizes, int n_in,
                              void* d_out, int out_size, void* d_ws,
                              size_t ws_size, hipStream_t stream) {
  (void)in_sizes; (void)n_in; (void)out_size; (void)ws_size;
  const float* C     = (const float*)d_in[0];
  const float* Q     = (const float*)d_in[1];
  const float* cmask = (const float*)d_in[2];
  const float* qmask = (const float*)d_in[3];
  const float* lp    = (const float*)d_in[4];
  float* out = (float*)d_out;

  char* w = (char*)d_ws;
  size_t o = 0;
  auto alloc = [&](size_t bytes) -> void* {
    void* p = (void*)(w + o);
    o += (bytes + 255) & ~(size_t)255;
    return p;
  };
  float* rowterm = (float*)alloc((size_t)B_ * LC_ * 4);
  float* colterm = (float*)alloc((size_t)B_ * LQ_ * 4);
  float* rowmax  = (float*)alloc((size_t)B_ * LC_ * 4);
  float* rowsum  = (float*)alloc((size_t)B_ * LC_ * 4);
  float* cpmax   = (float*)alloc((size_t)B_ * NBLK * LQ_ * 4);
  float* cpsum   = (float*)alloc((size_t)B_ * NBLK * LQ_ * 4);
  float* colmax  = (float*)alloc((size_t)B_ * LQ_ * 4);
  float* colsum  = (float*)alloc((size_t)B_ * LQ_ * 4);
  _Float16* CW   = (_Float16*)alloc((size_t)B_ * LC_ * H_ * 2);
  _Float16* QT16 = (_Float16*)alloc((size_t)B_ * LQ_ * H_ * 2);
  _Float16* Q16  = (_Float16*)alloc((size_t)B_ * H_ * LQ_ * 2);
  _Float16* Tt   = (_Float16*)alloc((size_t)B_ * H_ * LQ_ * 2);

  k_prep_c<<<B_ * (LC_ / 64), 256, 0, stream>>>(C, lp, CW, rowterm);
  k_prep_q<<<B_, 256, 0, stream>>>(Q, lp, QT16, Q16, colterm);
  k_stats<<<B_ * NBLK, 256, 0, stream>>>(CW, QT16, rowterm, colterm, cmask,
                                         qmask, rowmax, rowsum, cpmax, cpsum);
  k_colreduce<<<B_, 256, 0, stream>>>(cpmax, cpsum, colmax, colsum);
  k_T<<<B_ * 4, 256, 0, stream>>>(C, CW, QT16, rowterm, colterm, cmask, colmax,
                                  colsum, Tt);
  k_out<<<B_ * NBLK, 256, 128 * 264 * sizeof(_Float16), stream>>>(
      C, CW, QT16, Q16, Tt, rowterm, colterm, qmask, rowmax, rowsum, out);
}
